// CSWinB_CrossAttn_17652315586758
// MI455X (gfx1250) — matrix operand order repaired and checked by reading, not compiled, not run on hardware
//
#include <hip/hip_runtime.h>
#include <hip/hip_bf16.h>

typedef __attribute__((ext_vector_type(16))) _Float16 v16h;
typedef __attribute__((ext_vector_type(8)))  float    v8f;

#define WMMA_F16(a, b, c) \
  __builtin_amdgcn_wmma_f32_16x16x32_f16(false, (a), false, (b), (short)0, (c), false, false)

union V16HCvt { uint4 u[2]; v16h v; };

// ---- WMMA operand loaders (wave32 layouts per CDNA5 ISA 7.12.2) ----

// A 16x32 f16 from row-major [16 x ld] block (ld even, base 16B-aligned, ld*2 % 16 == 0).
// lane 0-15: M=lane, K = 0..7,16..23 ; lane 16-31: M=lane-16, K = 8..15,24..31
// Per lane: two contiguous 16B runs -> two b128 loads.
__device__ __forceinline__ v16h load_A16(const _Float16* __restrict__ base, int ld) {
  const int lane = threadIdx.x & 31;
  const int m = lane & 15;
  const int kb = (lane >> 4) << 3;
  const _Float16* p = base + (size_t)m * ld + kb;
  V16HCvt c;
  c.u[0] = *(const uint4*)(p);
  c.u[1] = *(const uint4*)(p + 16);
  return c.v;
}

// B 32x16 f16 from row-major [32(K) x ld(N...)] block (strided gather, unavoidable).
// lane 0-15: N=lane, K=0..7,16..23 ; lane 16-31: N=lane-16, K=8..15,24..31
__device__ __forceinline__ v16h load_B16(const _Float16* __restrict__ base, int ld) {
  const int lane = threadIdx.x & 31;
  const int n = lane & 15;
  const int kb = (lane >> 4) << 3;
  v16h b;
#pragma unroll
  for (int t = 0; t < 16; ++t) b[t] = base[(size_t)(kb + t + ((t >> 3) << 3)) * ld + n];
  return b;
}

// B 32x16 f16 where logical B = transpose of a row-major [16(N) x ld(K...)] block:
// element (K=k, N=n) = base[n*ld + k]. Per-lane: two 16B runs (K=kb..kb+7, kb+16..kb+23) -> two b128 loads.
__device__ __forceinline__ v16h load_Bt16(const _Float16* __restrict__ base, int ld) {
  const int lane = threadIdx.x & 31;
  const int n = lane & 15;
  const int kb = (lane >> 4) << 3;
  const _Float16* p = base + (size_t)n * ld + kb;
  V16HCvt c;
  c.u[0] = *(const uint4*)(p);
  c.u[1] = *(const uint4*)(p + 16);
  return c.v;
}

// Async memory -> LDS copy of 64 contiguous bytes (4 x b128), tracked by ASYNCcnt.
// lds = low 32 bits of the flat shared pointer (= LDS byte offset; aperture bits
// are entirely in addr[63:32] per the CDNA5 aperture check).
__device__ __forceinline__ void async_copy64(unsigned int lds, const _Float16* gsrc) {
  unsigned long long ga = (unsigned long long)(size_t)gsrc;
  asm volatile(
      "global_load_async_to_lds_b128 %0, %1, off\n\t"
      "global_load_async_to_lds_b128 %0, %1, off offset:16\n\t"
      "global_load_async_to_lds_b128 %0, %1, off offset:32\n\t"
      "global_load_async_to_lds_b128 %0, %1, off offset:48"
      :: "v"(lds), "v"(ga) : "memory");
}

// ---- Kernel 0: f32 -> f16 conversion ----
__global__ __launch_bounds__(256) void cvt_f32_to_f16(const float* __restrict__ src,
                                                      _Float16* __restrict__ dst, int n) {
  int i = blockIdx.x * blockDim.x + threadIdx.x;
  const int stride = gridDim.x * blockDim.x;
  for (; i < n; i += stride) dst[i] = (_Float16)src[i];
}

// ---- Kernel 1: generic f16 GEMM, f16 output. C[M,N] = A[M,K] * B[K,N] ----
// Block = 128 threads = 4 waves; each wave computes one 16x32 output tile.
__global__ __launch_bounds__(128) void gemm_f16f16(const _Float16* __restrict__ A,
                                                   const _Float16* __restrict__ B,
                                                   _Float16* __restrict__ C,
                                                   int K, int N) {
  const int lane = threadIdx.x & 31;
  const int wave = threadIdx.x >> 5;
  const int arow = (blockIdx.x * 4 + wave) * 16;
  const int ncol = blockIdx.y * 32;
  v8f acc0 = {}, acc1 = {};
  for (int ks = 0; ks < K; ks += 32) {
    v16h a  = load_A16(A + (size_t)arow * K + ks, K);
    v16h b0 = load_B16(B + (size_t)ks * N + ncol, N);
    v16h b1 = load_B16(B + (size_t)ks * N + ncol + 16, N);
    acc0 = WMMA_F16(a, b0, acc0);
    acc1 = WMMA_F16(a, b1, acc1);
  }
  const int n = lane & 15;
  const int mb = (lane >> 4) << 3;
#pragma unroll
  for (int r = 0; r < 8; ++r) {
    C[(size_t)(arow + mb + r) * N + ncol + n]      = (_Float16)acc0[r];
    C[(size_t)(arow + mb + r) * N + ncol + 16 + n] = (_Float16)acc1[r];
  }
}

// ---- Kernel 3: proj GEMM f16 x f16 -> f32 + bias ----
__global__ __launch_bounds__(128) void gemm_proj(const _Float16* __restrict__ A,
                                                 const _Float16* __restrict__ B,
                                                 const float* __restrict__ bias,
                                                 float* __restrict__ C,
                                                 int K, int N) {
  const int lane = threadIdx.x & 31;
  const int wave = threadIdx.x >> 5;
  const int arow = (blockIdx.x * 4 + wave) * 16;
  const int ncol = blockIdx.y * 32;
  v8f acc0 = {}, acc1 = {};
  for (int ks = 0; ks < K; ks += 32) {
    v16h a  = load_A16(A + (size_t)arow * K + ks, K);
    v16h b0 = load_B16(B + (size_t)ks * N + ncol, N);
    v16h b1 = load_B16(B + (size_t)ks * N + ncol + 16, N);
    acc0 = WMMA_F16(a, b0, acc0);
    acc1 = WMMA_F16(a, b1, acc1);
  }
  const int n = lane & 15;
  const int mb = (lane >> 4) << 3;
  const float bi0 = bias[ncol + n];
  const float bi1 = bias[ncol + 16 + n];
#pragma unroll
  for (int r = 0; r < 8; ++r) {
    C[(size_t)(arow + mb + r) * N + ncol + n]      = acc0[r] + bi0;
    C[(size_t)(arow + mb + r) * N + ncol + 16 + n] = acc1[r] + bi1;
  }
}

// ---- Kernel 2: windowed attention + fused depthwise-3x3 LePE ----
// One block = one (batch, branch, window, head). 128 threads = 4 waves.
// Window = 512 tokens, head dim = 32. Flash-style online softmax per 16-row Q tile.
__global__ __launch_bounds__(128) void attn_lepe(const _Float16* __restrict__ qkv,
                                                 const float* __restrict__ conv_ws,
                                                 const float* __restrict__ conv_bs,
                                                 _Float16* __restrict__ cat) {
  __shared__ _Float16 Klds[512 * 32];   // 32 KB
  __shared__ _Float16 Vlds[512 * 32];   // 32 KB
  __shared__ _Float16 Plds[4][16 * 32]; //  4 KB

  const int lane = threadIdx.x & 31;
  const int wave = threadIdx.x >> 5;

  int idx = blockIdx.x;
  const int bb = idx >> 7;            // batch 0..7
  int rem = idx & 127;
  const int br = rem >> 5;            // branch 0..3
  rem &= 31;
  const int w = rem >> 2;             // window 0..7
  const int h = rem & 3;              // head 0..3

  // branch channel slices: br0 q0/k0, br1 q1/k1, br2 q1/k0, br3 q0/k1 (v follows k)
  int qc, kc;
  switch (br) {
    case 0:  qc = 0;   kc = 0;   break;
    case 1:  qc = 128; kc = 128; break;
    case 2:  qc = 128; kc = 0;   break;
    default: qc = 0;   kc = 128; break;
  }
  const int qcol = qc + h * 32;             // q columns 0..255
  const int kcol = 256 + kc + h * 32;       // k columns 256..511
  const int vcol = 512 + kc + h * 32;       // v columns 512..767
  const bool vertical = (br == 0) || (br == 2);  // H_sp=64,W_sp=8 else H_sp=8,W_sp=64
  const int Wsp = vertical ? 8 : 64;
  const int Hsp = vertical ? 64 : 8;

  // stage K, V tiles (512 tokens x 32 channels) into LDS via async DMA (ASYNCcnt)
  for (int t = threadIdx.x; t < 512; t += 128) {
    const int L = vertical ? (((t >> 3) << 6) + (w << 3) + (t & 7)) : ((w << 9) + t);
    const _Float16* row = qkv + (size_t)(bb * 4096 + L) * 768;
    async_copy64((unsigned int)(size_t)(Klds + t * 32), row + kcol);
    async_copy64((unsigned int)(size_t)(Vlds + t * 32), row + vcol);
  }
  asm volatile("s_wait_asynccnt 0" ::: "memory");
  __syncthreads();

  // LePE weights: channel is fixed per (lane, nt)
  float wgt[2][9], lbias[2];
#pragma unroll
  for (int nt = 0; nt < 2; ++nt) {
    const int cg = h * 32 + nt * 16 + (lane & 15);
    const float* wp = conv_ws + ((size_t)br * 128 + cg) * 9;
#pragma unroll
    for (int u = 0; u < 9; ++u) wgt[nt][u] = wp[u];
    lbias[nt] = conv_bs[br * 128 + cg];
  }

  const float scale = 0.17677669529663687f;  // 1/sqrt(32)
  const int m = lane & 15;
  const int kb = (lane >> 4) << 3;
  const int nn = lane & 15;
  const int mb = (lane >> 4) << 3;

  for (int qt = wave; qt < 32; qt += 4) {
    // load Q A-tile (16 rows x 32 channels) straight from global via window map
    const int tq0 = qt * 16 + m;
    const int Lq0 = vertical ? (((tq0 >> 3) << 6) + (w << 3) + (tq0 & 7)) : ((w << 9) + tq0);
    const _Float16* qr = qkv + (size_t)(bb * 4096 + Lq0) * 768 + qcol + kb;
    V16HCvt qc2;
    qc2.u[0] = *(const uint4*)(qr);
    qc2.u[1] = *(const uint4*)(qr + 16);
    const v16h qa = qc2.v;

    float runmax[8], runsum[8];
#pragma unroll
    for (int r = 0; r < 8; ++r) { runmax[r] = -1e30f; runsum[r] = 0.f; }
    v8f oacc[2] = {{}, {}};

    for (int kt = 0; kt < 16; ++kt) {
      // S tile 16x32 = Q (16x32) * K^T
      v8f s0 = {}, s1 = {};
      v16h bk0 = load_Bt16(Klds + (size_t)(kt * 32) * 32, 32);
      v16h bk1 = load_Bt16(Klds + (size_t)(kt * 32 + 16) * 32, 32);
      s0 = WMMA_F16(qa, bk0, s0);
      s1 = WMMA_F16(qa, bk1, s1);

#pragma unroll
      for (int r = 0; r < 8; ++r) {
        s0[r] *= scale; s1[r] *= scale;
        // row max over 32 cols: butterfly within the 16-lane half holding this row
        float v = fmaxf(s0[r], s1[r]);
#pragma unroll
        for (int off = 8; off >= 1; off >>= 1) v = fmaxf(v, __shfl_xor(v, off, 32));
        const float mnew = fmaxf(runmax[r], v);
        const float p0 = __expf(s0[r] - mnew);
        const float p1 = __expf(s1[r] - mnew);
        s0[r] = p0; s1[r] = p1;
        float rs = p0 + p1;
#pragma unroll
        for (int off = 8; off >= 1; off >>= 1) rs += __shfl_xor(rs, off, 32);
        const float corr = __expf(runmax[r] - mnew);
        runsum[r] = runsum[r] * corr + rs;
        runmax[r] = mnew;
        oacc[0][r] *= corr; oacc[1][r] *= corr;
      }

      // re-layout P: C-layout regs -> LDS -> A-layout regs
      _Float16* P = &Plds[wave][0];
#pragma unroll
      for (int r = 0; r < 8; ++r) {
        P[(mb + r) * 32 + nn]      = (_Float16)s0[r];
        P[(mb + r) * 32 + 16 + nn] = (_Float16)s1[r];
      }
      asm volatile("s_wait_dscnt 0" ::: "memory");
      const _Float16* pr = P + m * 32 + kb;
      V16HCvt pc;
      pc.u[0] = *(const uint4*)(pr);
      pc.u[1] = *(const uint4*)(pr + 16);
      const v16h pa = pc.v;

      // O += P (16x32) * V (32x32)
      v16h bv0 = load_B16(Vlds + (size_t)(kt * 32) * 32, 32);
      v16h bv1 = load_B16(Vlds + (size_t)(kt * 32) * 32 + 16, 32);
      oacc[0] = WMMA_F16(pa, bv0, oacc[0]);
      oacc[1] = WMMA_F16(pa, bv1, oacc[1]);
    }

    // epilogue: normalize, add depthwise-3x3 LePE (from LDS-resident V), store f16
#pragma unroll
    for (int nt = 0; nt < 2; ++nt) {
      const int d = nt * 16 + nn;
#pragma unroll
      for (int r = 0; r < 8; ++r) {
        const int tq = qt * 16 + mb + r;
        const int i = vertical ? (tq >> 3) : (tq >> 6);
        const int j = vertical ? (tq & 7) : (tq & 63);
        float lepe = lbias[nt];
#pragma unroll
        for (int di = -1; di <= 1; ++di) {
          const int ii = i + di;
          if (ii < 0 || ii >= Hsp) continue;
#pragma unroll
          for (int dj = -1; dj <= 1; ++dj) {
            const int jj = j + dj;
            if (jj < 0 || jj >= Wsp) continue;
            lepe += wgt[nt][(di + 1) * 3 + (dj + 1)] *
                    (float)Vlds[(ii * Wsp + jj) * 32 + d];
          }
        }
        const float val = oacc[nt][r] / runsum[r] + lepe;
        const int Lq = vertical ? (((tq >> 3) << 6) + (w << 3) + (tq & 7)) : ((w << 9) + tq);
        cat[(size_t)(bb * 4096 + Lq) * 512 + br * 128 + h * 32 + d] = (_Float16)val;
      }
    }
  }
}

extern "C" void kernel_launch(void* const* d_in, const int* in_sizes, int n_in,
                              void* d_out, int out_size, void* d_ws, size_t ws_size,
                              hipStream_t stream) {
  (void)in_sizes; (void)n_in; (void)out_size; (void)ws_size;
  const float* x       = (const float*)d_in[0];
  const float* qkv_w   = (const float*)d_in[1];
  const float* proj_w  = (const float*)d_in[2];
  const float* proj_b  = (const float*)d_in[3];
  const float* conv_ws = (const float*)d_in[4];
  const float* conv_bs = (const float*)d_in[5];
  float* out = (float*)d_out;

  char* ws = (char*)d_ws;
  size_t off = 0;
  auto carve = [&](size_t bytes) -> char* {
    char* p = ws + off;
    off = (off + bytes + 255) & ~(size_t)255;
    return p;
  };
  _Float16* xh     = (_Float16*)carve((size_t)32768 * 256 * 2);  // x in f16
  _Float16* qkvwh  = (_Float16*)carve((size_t)256 * 768 * 2);    // qkv_w f16
  _Float16* projwh = (_Float16*)carve((size_t)512 * 256 * 2);    // proj_w f16
  _Float16* qkvh   = (_Float16*)carve((size_t)32768 * 768 * 2);  // qkv activations f16
  _Float16* cath   = (_Float16*)carve((size_t)32768 * 512 * 2);  // concat attn out f16

  cvt_f32_to_f16<<<2048, 256, 0, stream>>>(x, xh, 32768 * 256);
  cvt_f32_to_f16<<<768, 256, 0, stream>>>(qkv_w, qkvwh, 256 * 768);
  cvt_f32_to_f16<<<512, 256, 0, stream>>>(proj_w, projwh, 512 * 256);

  // QKV GEMM: (32768 x 256) * (256 x 768) -> f16
  gemm_f16f16<<<dim3(512, 24), 128, 0, stream>>>(xh, qkvwh, qkvh, 256, 768);

  // Attention + LePE: 8 batches x 4 branches x 8 windows x 4 heads = 1024 blocks
  attn_lepe<<<1024, 128, 0, stream>>>(qkvh, conv_ws, conv_bs, cath);

  // Projection: (32768 x 512) * (512 x 256) + bias -> f32 out
  gemm_proj<<<dim3(512, 8), 128, 0, stream>>>(cath, projwh, proj_b, out, 512, 256);
}
